// NodeAlignerGNN_60370060313338
// MI455X (gfx1250) — matrix operand order repaired and checked by reading, not compiled, not run on hardware
//
#include <hip/hip_runtime.h>
#include <math.h>

typedef __attribute__((ext_vector_type(16))) _Float16 v16h;
typedef __attribute__((ext_vector_type(4)))  _Float16 v4h;
typedef __attribute__((ext_vector_type(8)))  float    v8f;

#define NEG_SLOPE 0.2f
#define EPSV 1e-16f
#define NEG_INF (-3.0e38f)

#if defined(__has_builtin) && __has_builtin(__builtin_amdgcn_sched_barrier)
#define SCHED_FENCE() __builtin_amdgcn_sched_barrier(0)
#else
#define SCHED_FENCE()
#endif

__device__ __forceinline__ float lrelu(float x) { return x > 0.0f ? x : NEG_SLOPE * x; }

__device__ __forceinline__ void atomicMaxF(float* addr, float val) {
    unsigned int* ua = (unsigned int*)addr;
    unsigned int old = __float_as_uint(*addr);
    while (__uint_as_float(old) < val) {
        unsigned int assumed = old;
        old = atomicCAS(ua, assumed, __float_as_uint(val));
        if (old == assumed) break;
    }
}

__device__ __forceinline__ v4h cvt4(float4 f) {
    v4h r;
    r[0] = (_Float16)f.x; r[1] = (_Float16)f.y;
    r[2] = (_Float16)f.z; r[3] = (_Float16)f.w;
    return r;
}

// ---------------------------------------------------------------------------
// WMMA GEMM: C[M x 128] = A[M x 128] * B[128 x 128]  (+bias[128], optional relu)
// f32 inputs -> f16 LDS tiles -> v_wmma_f32_16x16x32_f16 (f32 accumulate).
// Block = 128 threads (4 waves); block tile = 64 rows x 64 cols; grid.y = 2.
// Staging: 8x global_load_b128 issued back-to-back (sched_barrier keeps the
// converts/selects out of the load cluster, so there's one loadcnt wait that
// lands AFTER the current step's WMMAs), branchless row clamp, packed v4h
// ds_store_b64. A fragment per ISA 16-bit A 16x32 layout; B staged k-major so
// each lane's fragment (lane = N column, 16 K values) is 16 strided reads down one LDS column.
// ---------------------------------------------------------------------------
__global__ __launch_bounds__(128)
void gemm128_wmma(const float* __restrict__ A, const float* __restrict__ B,
                  float* __restrict__ C, int M,
                  const float* __restrict__ bias, int relu) {
    __shared__ _Float16 As[64][32];
    __shared__ _Float16 Bs[32][64];
    const int t    = threadIdx.x;
    const int lane = t & 31;
    const int wave = t >> 5;
    const int row0 = blockIdx.x * 64;
    const int n0   = blockIdx.y * 64;
    const int m    = lane & 15;
    const int half = lane >> 4;

    v8f acc[4];
#pragma unroll
    for (int i = 0; i < 4; ++i) acc[i] = (v8f){0,0,0,0,0,0,0,0};

    float4 Ar[4], Br[4];

    auto loadStage = [&](int k0) {
#pragma unroll
        for (int i = 0; i < 4; ++i) {
            int v  = i * 128 + t;
            int r  = v >> 3, c4 = v & 7;        // A: 8 float4 per 32-wide row
            int gr = row0 + r;
            int sr = gr < M ? gr : (M - 1);     // branchless clamp, keep loads batched
            Ar[i] = ((const float4*)(A + (long long)sr * 128 + k0))[c4];
            int kk = v >> 4, n4 = v & 15;       // B: 16 float4 per 64-wide row
            Br[i] = ((const float4*)(B + (k0 + kk) * 128 + n0))[n4];
        }
        SCHED_FENCE();  // pin the 8 b128 loads together; converts stay below
    };

    auto storeStage = [&]() {
#pragma unroll
        for (int i = 0; i < 4; ++i) {
            int v = i * 128 + t;
            int r = v >> 3, c4 = v & 7;
            int gr = row0 + r;
            float4 a = Ar[i];
            if (gr >= M) a = make_float4(0.f, 0.f, 0.f, 0.f);
            *(v4h*)&As[r][c4 * 4] = cvt4(a);
            int kk = v >> 4, n4 = v & 15;
            *(v4h*)&Bs[kk][n4 * 4] = cvt4(Br[i]);
        }
        SCHED_FENCE();
    };

    loadStage(0);
#pragma unroll
    for (int step = 0; step < 4; ++step) {
        __syncthreads();                         // LDS from prev step fully consumed
        storeStage();
        __syncthreads();
        if (step < 3) loadStage((step + 1) * 32);  // loads drain behind the WMMAs below

        v16h afrag;
        const _Float16* arow = &As[wave * 16 + m][0];
#pragma unroll
        for (int v = 0; v < 8; ++v) {
            int kb = (v < 4 ? v * 2 : 16 + (v - 4) * 2) + half * 8;
            afrag[2 * v]     = arow[kb];
            afrag[2 * v + 1] = arow[kb + 1];
        }
#pragma unroll
        for (int tt = 0; tt < 4; ++tt) {
            v16h bfrag;
#pragma unroll
            for (int s = 0; s < 16; ++s) {
                int kb = (s < 8 ? s : s + 8) + half * 8;
                bfrag[s] = Bs[kb][tt * 16 + m];
            }
            acc[tt] = __builtin_amdgcn_wmma_f32_16x16x32_f16(
                false, afrag, false, bfrag, (short)0, acc[tt], false, false);
        }
    }

    // C/D layout: lane l -> n = l&15, row = r + 8*(l>>4)
#pragma unroll
    for (int tt = 0; tt < 4; ++tt) {
#pragma unroll
        for (int r = 0; r < 8; ++r) {
            int mm = row0 + wave * 16 + r + half * 8;
            if (mm < M) {
                int nn = n0 + tt * 16 + m;
                float v = acc[tt][r];
                if (bias) v += bias[nn];
                if (relu) v = v > 0.0f ? v : 0.0f;
                C[(long long)mm * 128 + nn] = v;
            }
        }
    }
}

// ---------------------------------------------------------------------------
// Per-node attention scores: one wave per node, 128 channels (float4/lane).
// ---------------------------------------------------------------------------
template <int H>
__global__ void att_scores(const float* __restrict__ h,
                           const float* __restrict__ att_src,
                           const float* __restrict__ att_dst,
                           float* __restrict__ asrc, float* __restrict__ adst,
                           int N) {
    int wid  = (int)((blockIdx.x * (long long)blockDim.x + threadIdx.x) >> 5);
    int lane = threadIdx.x & 31;
    if (wid >= N) return;
    float4 v = ((const float4*)(h + (long long)wid * 128))[lane];
    float4 a = ((const float4*)att_src)[lane];
    float4 d = ((const float4*)att_dst)[lane];
    float ps = v.x * a.x + v.y * a.y + v.z * a.z + v.w * a.w;
    float pd = v.x * d.x + v.y * d.y + v.z * d.z + v.w * d.w;
    constexpr int lanesPerHead = 32 / H;  // 16 (H=2) or 32 (H=1)
#pragma unroll
    for (int off = lanesPerHead >> 1; off > 0; off >>= 1) {
        ps += __shfl_xor(ps, off, 32);
        pd += __shfl_xor(pd, off, 32);
    }
    if ((lane & (lanesPerHead - 1)) == 0) {
        int head = lane / lanesPerHead;
        asrc[(long long)wid * H + head] = ps;
        adst[(long long)wid * H + head] = pd;
    }
}

__global__ void fill_f32(float* p, float v, long long n) {
    long long i = blockIdx.x * (long long)blockDim.x + threadIdx.x;
    if (i < n) p[i] = v;
}

__global__ void fill_bias128(float* __restrict__ out, const float* __restrict__ bias, int N) {
    long long i = blockIdx.x * (long long)blockDim.x + threadIdx.x;
    if (i < (long long)N * 128) out[i] = bias[i & 127];
}

// edge pass 1: segment max over destination (thread per edge*head)
template <int H>
__global__ void edge_max(const long long* __restrict__ srcIdx,
                         const long long* __restrict__ dstIdx,
                         int E, int N,
                         const float* __restrict__ asrc, const float* __restrict__ adst,
                         float* __restrict__ mbuf) {
    long long t = blockIdx.x * (long long)blockDim.x + threadIdx.x;
    long long total = (long long)(E + N) * H;
    if (t >= total) return;
    long long e = t / H; int h = (int)(t - e * H);
    long long s, d;
    if (e < E) { s = srcIdx[e]; d = dstIdx[e]; } else { s = d = e - E; }
    float val = lrelu(asrc[s * H + h] + adst[d * H + h]);
    atomicMaxF(&mbuf[d * H + h], val);
}

// edge pass 2: ex = exp(e - max); store per-edge, atomic-sum per dst
template <int H>
__global__ void edge_expsum(const long long* __restrict__ srcIdx,
                            const long long* __restrict__ dstIdx,
                            int E, int N,
                            const float* __restrict__ asrc, const float* __restrict__ adst,
                            const float* __restrict__ mbuf,
                            float* __restrict__ exbuf, float* __restrict__ sbuf) {
    long long t = blockIdx.x * (long long)blockDim.x + threadIdx.x;
    long long total = (long long)(E + N) * H;
    if (t >= total) return;
    long long e = t / H; int h = (int)(t - e * H);
    long long s, d;
    if (e < E) { s = srcIdx[e]; d = dstIdx[e]; } else { s = d = e - E; }
    float val = lrelu(asrc[s * H + h] + adst[d * H + h]);
    float ex  = expf(val - mbuf[d * H + h]);
    exbuf[t] = ex;
    atomicAdd(&sbuf[d * H + h], ex);
}

// edge pass 3: out[dst] += h[src] * alpha   (wave per edge, float4 per lane,
// scatter via f32 atomics that stay resident in the 192MB L2)
template <int H>
__global__ void edge_aggregate(const long long* __restrict__ srcIdx,
                               const long long* __restrict__ dstIdx,
                               int E, int N,
                               const float* __restrict__ exbuf,
                               const float* __restrict__ sbuf,
                               const float* __restrict__ hin,
                               float* __restrict__ out) {
    long long wid = (blockIdx.x * (long long)blockDim.x + threadIdx.x) >> 5;
    int lane = threadIdx.x & 31;
    long long total = (long long)E + N;
    if (wid >= total) return;
    long long s, d;
    if (wid < E) { s = srcIdx[wid]; d = dstIdx[wid]; } else { s = d = wid - E; }
    constexpr int C = 128 / H;
    int c0 = lane * 4;
    int head = c0 / C;
    float alpha = exbuf[wid * H + head] / (sbuf[d * H + head] + EPSV);
    float4 hv = ((const float4*)(hin + s * 128))[lane];
    float* op = out + d * 128 + c0;
    atomicAdd(op + 0, hv.x * alpha);
    atomicAdd(op + 1, hv.y * alpha);
    atomicAdd(op + 2, hv.z * alpha);
    atomicAdd(op + 3, hv.w * alpha);
}

// cq[j] = bq1[j] + sum_k q[k] * Wq1[128+k, j]   (q is row-constant -> fold once)
__global__ void query_const(const float* __restrict__ q, const float* __restrict__ Wq1,
                            const float* __restrict__ bq1, float* __restrict__ cq) {
    int j = threadIdx.x;  // 128 threads
    float acc = bq1[j];
    for (int k = 0; k < 384; ++k) acc += q[k] * Wq1[(128 + k) * 128 + j];
    cq[j] = acc;
}

// scores[n] = dot(Z[n,:], w) + b   (wave per row)
__global__ void row_dot128(const float* __restrict__ Z, const float* __restrict__ w,
                           const float* __restrict__ b, float* __restrict__ s, int N) {
    int wid  = (int)((blockIdx.x * (long long)blockDim.x + threadIdx.x) >> 5);
    int lane = threadIdx.x & 31;
    if (wid >= N) return;
    float4 zv = ((const float4*)(Z + (long long)wid * 128))[lane];
    float4 wv = ((const float4*)w)[lane];
    float p = zv.x * wv.x + zv.y * wv.y + zv.z * wv.z + zv.w * wv.w;
#pragma unroll
    for (int off = 16; off > 0; off >>= 1) p += __shfl_xor(p, off, 32);
    if (lane == 0) s[wid] = p + b[0];
}

__global__ void reduce_max(const float* __restrict__ s, int N, float* __restrict__ gmax) {
    int i = blockIdx.x * blockDim.x + threadIdx.x;
    float v = (i < N) ? s[i] : NEG_INF;
#pragma unroll
    for (int off = 16; off > 0; off >>= 1) v = fmaxf(v, __shfl_xor(v, off, 32));
    if ((threadIdx.x & 31) == 0) atomicMaxF(gmax, v);
}

__global__ void reduce_sumexp(const float* __restrict__ s, int N,
                              const float* __restrict__ gmax, float* __restrict__ gsum) {
    int i = blockIdx.x * blockDim.x + threadIdx.x;
    float v = (i < N) ? expf(s[i] - *gmax) : 0.0f;
#pragma unroll
    for (int off = 16; off > 0; off >>= 1) v += __shfl_xor(v, off, 32);
    if ((threadIdx.x & 31) == 0) atomicAdd(gsum, v);
}

__global__ void softmax_write(const float* __restrict__ s, int N,
                              const float* __restrict__ gmax, const float* __restrict__ gsum,
                              float* __restrict__ out) {
    int i = blockIdx.x * blockDim.x + threadIdx.x;
    if (i < N) out[i] = expf(s[i] - *gmax) / (*gsum);
}

// ---------------------------------------------------------------------------
extern "C" void kernel_launch(void* const* d_in, const int* in_sizes, int n_in,
                              void* d_out, int out_size, void* d_ws, size_t ws_size,
                              hipStream_t stream) {
    const float*     x    = (const float*)d_in[0];
    const long long* edge = (const long long*)d_in[1];  // int64 [2, E]
    const float*     q    = (const float*)d_in[3];
    const float*     W1   = (const float*)d_in[4];
    const float*     as1  = (const float*)d_in[5];
    const float*     ad1  = (const float*)d_in[6];
    const float*     b1   = (const float*)d_in[7];
    const float*     W2   = (const float*)d_in[8];
    const float*     as2  = (const float*)d_in[9];
    const float*     ad2  = (const float*)d_in[10];
    const float*     b2   = (const float*)d_in[11];
    const float*     Wq1  = (const float*)d_in[12];
    const float*     bq1  = (const float*)d_in[13];
    const float*     Wq2  = (const float*)d_in[14];
    const float*     bq2  = (const float*)d_in[15];

    const int N = in_sizes[0] / 128;
    const int E = in_sizes[1] / 2;
    const long long* srcI = edge;
    const long long* dstI = edge + E;

    // workspace layout (floats)
    float* ws = (float*)d_ws;
    long long off = 0;
    float* bufA   = ws + off; off += (long long)N * 128;  // h1_lin / h2_lin / z(relu)
    float* bufB   = ws + off; off += (long long)N * 128;  // layer-1 aggregated output
    float* asrc   = ws + off; off += (long long)N * 2;
    float* adst   = ws + off; off += (long long)N * 2;
    float* mbuf   = ws + off; off += (long long)N * 2;
    float* sbuf   = ws + off; off += (long long)N * 2;
    float* exbuf  = ws + off; off += (long long)(E + N) * 2;
    float* scores = ws + off; off += N;
    float* cq     = ws + off; off += 128;
    float* scal   = ws + off; off += 2;                   // [gmax, gsum]
    float* hout   = (float*)d_out;                        // [N,128] then softmax [N]

    const dim3 gblk(128), blk256(256);
    const dim3 ggrid((N + 63) / 64, 2);
    const int nodeWaveBlocks = (int)(((long long)N * 32 + 255) / 256);
    auto edgeBlocks = [&](int H) { return (int)(((long long)(E + N) * H + 255) / 256); };
    const int aggBlocks    = (int)(((long long)(E + N) * 32 + 255) / 256);
    const int fillBlocksNH = (int)(((long long)N * 128 + 255) / 256);

    // ---- GAT layer 1 (H=2, C=64) ----
    gemm128_wmma<<<ggrid, gblk, 0, stream>>>(x, W1, bufA, N, nullptr, 0);
    att_scores<2><<<nodeWaveBlocks, blk256, 0, stream>>>(bufA, as1, ad1, asrc, adst, N);
    fill_f32<<<(int)(((long long)N * 2 + 255) / 256), blk256, 0, stream>>>(mbuf, NEG_INF, (long long)N * 2);
    fill_f32<<<(int)(((long long)N * 2 + 255) / 256), blk256, 0, stream>>>(sbuf, 0.0f, (long long)N * 2);
    edge_max<2><<<edgeBlocks(2), blk256, 0, stream>>>(srcI, dstI, E, N, asrc, adst, mbuf);
    edge_expsum<2><<<edgeBlocks(2), blk256, 0, stream>>>(srcI, dstI, E, N, asrc, adst, mbuf, exbuf, sbuf);
    fill_bias128<<<fillBlocksNH, blk256, 0, stream>>>(bufB, b1, N);
    edge_aggregate<2><<<aggBlocks, blk256, 0, stream>>>(srcI, dstI, E, N, exbuf, sbuf, bufA, bufB);

    // ---- GAT layer 2 (H=1, C=128), output -> d_out[:N*128] ----
    gemm128_wmma<<<ggrid, gblk, 0, stream>>>(bufB, W2, bufA, N, nullptr, 0);
    att_scores<1><<<nodeWaveBlocks, blk256, 0, stream>>>(bufA, as2, ad2, asrc, adst, N);
    fill_f32<<<(int)(((long long)N + 255) / 256), blk256, 0, stream>>>(mbuf, NEG_INF, N);
    fill_f32<<<(int)(((long long)N + 255) / 256), blk256, 0, stream>>>(sbuf, 0.0f, N);
    edge_max<1><<<edgeBlocks(1), blk256, 0, stream>>>(srcI, dstI, E, N, asrc, adst, mbuf);
    edge_expsum<1><<<edgeBlocks(1), blk256, 0, stream>>>(srcI, dstI, E, N, asrc, adst, mbuf, exbuf, sbuf);
    fill_bias128<<<fillBlocksNH, blk256, 0, stream>>>(hout, b2, N);
    edge_aggregate<1><<<aggBlocks, blk256, 0, stream>>>(srcI, dstI, E, N, exbuf, sbuf, bufA, hout);

    // ---- Query MLP (q folded into constant bias) + softmax ----
    query_const<<<1, 128, 0, stream>>>(q, Wq1, bq1, cq);
    gemm128_wmma<<<ggrid, gblk, 0, stream>>>(hout, Wq1, bufA, N, cq, 1);  // relu(h2@Wq1[:128] + cq)
    row_dot128<<<nodeWaveBlocks, blk256, 0, stream>>>(bufA, Wq2, bq2, scores, N);
    fill_f32<<<1, 1, 0, stream>>>(scal, NEG_INF, 1);
    fill_f32<<<1, 1, 0, stream>>>(scal + 1, 0.0f, 1);
    reduce_max<<<(N + 255) / 256, blk256, 0, stream>>>(scores, N, scal);
    reduce_sumexp<<<(N + 255) / 256, blk256, 0, stream>>>(scores, N, scal, scal + 1);
    softmax_write<<<(N + 255) / 256, blk256, 0, stream>>>(scores, N, scal, scal + 1,
                                                          hout + (long long)N * 128);
}
